// CBM_24550033064414
// MI455X (gfx1250) — compile-verified, matrix operand order checked by reading, not run on hardware
//
#include <hip/hip_runtime.h>
#include <hip/hip_bf16.h>

// MI455X (gfx1250) implementation.
// Heavy GEMMs: v_wmma_f32_16x16x32_bf16, 8 waves -> 32x64 block tile, with
// double-buffered LDS staging fed by global_load_async_to_lds_b128 (ASYNCcnt).
// Scoring GEMM (4096x50000x512) is fused with an in-register per-row top-32.

#define DD   512
#define KK   32
#define NN   50000
#define BB   4096
#define HH   8
#define FFD  2048
#define LL   6

typedef __bf16 bf16;
typedef __bf16  v16bf __attribute__((ext_vector_type(16)));
typedef __bf16  v8bf  __attribute__((ext_vector_type(8)));
typedef float   v8f   __attribute__((ext_vector_type(8)));

// ---------------------------------------------------------------- helpers ---
__device__ __forceinline__ v16bf combine16(v8bf lo, v8bf hi) {
    v16bf r;
#pragma unroll
    for (int i = 0; i < 8; ++i) { r[i] = lo[i]; r[i + 8] = hi[i]; }
    return r;
}

__device__ __forceinline__ float wave_sum(float v) {
#pragma unroll
    for (int d = 16; d; d >>= 1) v += __shfl_xor(v, d);
    return v;
}
__device__ __forceinline__ float wave_max(float v) {
#pragma unroll
    for (int d = 16; d; d >>= 1) v = fmaxf(v, __shfl_xor(v, d));
    return v;
}
__device__ __forceinline__ float block_sum256(float v, float* red) {
    const int lane = threadIdx.x & 31, wave = threadIdx.x >> 5;
    v = wave_sum(v);
    __syncthreads();
    if (lane == 0) red[wave] = v;
    __syncthreads();
    float s = 0.f;
#pragma unroll
    for (int i = 0; i < 8; ++i) s += red[i];
    return s;
}

// Async copy of 16 bytes global -> LDS (per lane). LDS address is a 32-bit
// byte offset (low 32 bits of the generic shared pointer, ISA 10.2 aperture
// truncation). Tracked by ASYNCcnt.
__device__ __forceinline__ void async_b128(unsigned lds_off, const void* g) {
    asm volatile("global_load_async_to_lds_b128 %0, %1, off"
                 :: "v"(lds_off), "v"((unsigned long long)(uintptr_t)g)
                 : "memory");
}

// --------------------------------------------------------------- convert ----
// n8 = element count / 8; all tensors here are multiples of 8 elements.
__global__ void cvt_f32_bf16(const float* __restrict__ src, bf16* __restrict__ dst, size_t n8) {
    for (size_t i = (size_t)blockIdx.x * blockDim.x + threadIdx.x; i < n8;
         i += (size_t)gridDim.x * blockDim.x) {
        const float4* s = (const float4*)(src + i * 8);
        float4 a = s[0], b = s[1];
        v8bf o;
        o[0] = (bf16)a.x; o[1] = (bf16)a.y; o[2] = (bf16)a.z; o[3] = (bf16)a.w;
        o[4] = (bf16)b.x; o[5] = (bf16)b.y; o[6] = (bf16)b.z; o[7] = (bf16)b.w;
        *(v8bf*)(dst + i * 8) = o;
    }
}

__global__ void init_t_kernel(const float* __restrict__ ce, float* __restrict__ t,
                              bf16* __restrict__ tb, size_t n8) {
    for (size_t i = (size_t)blockIdx.x * blockDim.x + threadIdx.x; i < n8;
         i += (size_t)gridDim.x * blockDim.x) {
        const size_t e0 = i * 8;
        const float4* s = (const float4*)(ce + (e0 & (DD - 1)));
        float4 a = s[0], b = s[1];
        ((float4*)(t + e0))[0] = a;
        ((float4*)(t + e0))[1] = b;
        v8bf o;
        o[0] = (bf16)a.x; o[1] = (bf16)a.y; o[2] = (bf16)a.z; o[3] = (bf16)a.w;
        o[4] = (bf16)b.x; o[5] = (bf16)b.y; o[6] = (bf16)b.z; o[7] = (bf16)b.w;
        *(v8bf*)(tb + e0) = o;
    }
}

// ------------------------------------------------------------ WMMA GEMM -----
// C[M,N] = A[M,K] @ W[N,K]^T + bias, optional relu. A,W bf16; C -> f32 and/or
// bf16. Block = 256 threads (8 waves). 32x64 block tile; K consumed in 128-wide
// panels staged into LDS (A: 8KB, W: 16KB) with double buffering (48KB total),
// filled by async-to-LDS loads, drained with s_wait_asynccnt. Last stage peeled
// so the steady-state loop is straight-line.
__global__ void gemm_wmma(const bf16* __restrict__ A, const bf16* __restrict__ W,
                          const float* __restrict__ bias,
                          float* __restrict__ Cf, bf16* __restrict__ Cb,
                          int M, int N, int Kd, int relu) {
    const int tid  = threadIdx.x;
    const int lane = tid & 31;
    const int wave = tid >> 5;
    const int mt = wave >> 2;          // 0..1
    const int nt = wave & 3;           // 0..3
    const int m0b = blockIdx.y * 32;
    const int n0b = blockIdx.x * 64;

    // stage layout (elements): [A 32x128 = 4096][W 64x128 = 8192], x2 buffers
    __shared__ bf16 smem[2 * 12288];
    const unsigned smem_base = (unsigned)(uintptr_t)(void*)&smem[0];

    const int nK = Kd >> 7;            // K panels of 128 (Kd in {512,2048})

    auto issue = [&](int kt) {
        const unsigned sb = smem_base + (unsigned)(kt & 1) * 24576u;
        const int k0 = kt << 7;
        // A panel: 512 chunks of 8 elems; 2 per thread
#pragma unroll
        for (int j = 0; j < 2; ++j) {
            const int c = j * 256 + tid;
            const int row = c >> 4, col = (c & 15) << 3;
            async_b128(sb + (unsigned)c * 16u,
                       A + (size_t)(m0b + row) * Kd + k0 + col);
        }
        // W panel: 1024 chunks; 4 per thread
#pragma unroll
        for (int j = 0; j < 4; ++j) {
            const int c = j * 256 + tid;
            const int row = c >> 4, col = (c & 15) << 3;
            async_b128(sb + 8192u + (unsigned)c * 16u,
                       W + (size_t)(n0b + row) * Kd + k0 + col);
        }
    };

    v8f c = {};
    const int r    = lane & 15;
    const int half = (lane & 16) ? 8 : 0;
    auto compute = [&](int kt) {
        const bf16* Ab = smem + (size_t)(kt & 1) * 12288;
        const bf16* Wb = Ab + 4096;
#pragma unroll
        for (int ks = 0; ks < 4; ++ks) {
            const bf16* ap = Ab + (mt * 16 + r) * 128 + ks * 32 + half;
            const bf16* wp = Wb + (nt * 16 + r) * 128 + ks * 32 + half;
            v16bf a = combine16(*(const v8bf*)ap, *(const v8bf*)(ap + 16));
            v16bf b = combine16(*(const v8bf*)wp, *(const v8bf*)(wp + 16));
            c = __builtin_amdgcn_wmma_f32_16x16x32_bf16(false, a, false, b,
                                                        (short)0, c, false, false);
        }
    };

    issue(0);
    for (int kt = 0; kt < nK - 1; ++kt) {
        issue(kt + 1);
        // 6 async instrs per wave per stage; loads retire in order, so
        // <=6 outstanding means the previous stage has fully landed.
        asm volatile("s_wait_asynccnt 6" ::: "memory");
        __syncthreads();
        compute(kt);
        __syncthreads();   // all waves done reading before buffer is re-filled
    }
    asm volatile("s_wait_asynccnt 0" ::: "memory");
    __syncthreads();
    compute(nK - 1);

    const int col   = n0b + nt * 16 + (lane & 15);
    const int rbase = m0b + mt * 16 + ((lane & 16) ? 8 : 0);
    const float bv  = bias ? bias[col] : 0.f;
#pragma unroll
    for (int rr = 0; rr < 8; ++rr) {
        float v = c[rr] + bv;
        if (relu) v = fmaxf(v, 0.f);
        size_t off = (size_t)(rbase + rr) * N + col;
        if (Cf) Cf[off] = v;
        if (Cb) Cb[off] = (bf16)v;
    }
}

// -------------------------------------------- scoring + top-32 + kv build ---
// One block per 16 batch rows. x tile staged in LDS; loop over N in chunks of
// 128 cols (8 waves x 16). Per-row top-32 lives in wave registers (lane j
// holds slot j); rare insertions via ballot + wave-argmin replace.
__global__ void score_topk_kv(const bf16* __restrict__ xb, const bf16* __restrict__ cfb,
                              const float* __restrict__ xf, const float* __restrict__ cff,
                              const float* __restrict__ type_emb,
                              const float* __restrict__ pos_emb,
                              bf16* __restrict__ kv) {
    const int lane = threadIdx.x & 31;
    const int wave = threadIdx.x >> 5;
    const int mb   = blockIdx.x * 16;

    __shared__ bf16  xa[16 * DD];          // 16 KB
    __shared__ float sc[16][128];          // 8 KB

    for (int i = threadIdx.x; i < 16 * DD; i += 256)
        xa[i] = xb[(size_t)(mb + (i >> 9)) * DD + (i & (DD - 1))];
    __syncthreads();

    float tv[2]; int ti[2]; float curmin[2];
    tv[0] = tv[1] = -3.0e38f; ti[0] = ti[1] = 0;
    curmin[0] = curmin[1] = -3.0e38f;

    const int nChunks = (NN + 127) / 128;
    for (int ch = 0; ch < nChunks; ++ch) {
        // ---- 16x16 score tile for this wave
        const int n0   = ch * 128 + wave * 16;
        const int r    = lane & 15;
        const int half = (lane & 16) ? 8 : 0;
        const int col  = n0 + r;
        const int ccl  = col < NN ? col : 0;
        const bf16* bp = cfb + (size_t)ccl * DD;
        v8f c = {};
        for (int k0 = 0; k0 < DD; k0 += 32) {
            const bf16* ap = &xa[r * DD + k0 + half];
            v16bf a = combine16(*(const v8bf*)ap, *(const v8bf*)(ap + 16));
            v16bf b = combine16(*(const v8bf*)(bp + k0 + half),
                                *(const v8bf*)(bp + k0 + 16 + half));
            c = __builtin_amdgcn_wmma_f32_16x16x32_bf16(false, a, false, b,
                                                        (short)0, c, false, false);
        }
        const int ccol = wave * 16 + (lane & 15);
        const int gc0  = ch * 128 + ccol;
        const int rb   = (lane & 16) ? 8 : 0;
#pragma unroll
        for (int rr = 0; rr < 8; ++rr)
            sc[rb + rr][ccol] = (gc0 < NN) ? c[rr] : -3.0e38f;
        __syncthreads();

        // ---- top-32 maintenance: this wave owns rows 2*wave, 2*wave+1
#pragma unroll
        for (int rr = 0; rr < 2; ++rr) {
            const int row = wave * 2 + rr;
#pragma unroll
            for (int t = 0; t < 4; ++t) {
                float s  = sc[row][t * 32 + lane];
                int   gc = ch * 128 + t * 32 + lane;
                unsigned msk = (unsigned)__ballot(s > curmin[rr]);
                while (msk) {
                    int l = __ffs(msk) - 1;
                    msk &= msk - 1;
                    float sv = __shfl(s, l);
                    int   si = __shfl(gc, l);
                    if (sv > curmin[rr]) {
                        float mv = tv[rr]; int mp = lane;
#pragma unroll
                        for (int d = 16; d; d >>= 1) {
                            float ov = __shfl_xor(mv, d);
                            int   op = __shfl_xor(mp, d);
                            if (ov < mv) { mv = ov; mp = op; }
                        }
                        if (lane == mp) { tv[rr] = sv; ti[rr] = si; }
                        float nm = tv[rr];
#pragma unroll
                        for (int d = 16; d; d >>= 1) nm = fminf(nm, __shfl_xor(nm, d));
                        curmin[rr] = nm;
                    }
                }
            }
        }
        __syncthreads();
    }

    // ---- sort descending (bitonic, 32 lanes), softmax, emit kv rows
#pragma unroll
    for (int rr = 0; rr < 2; ++rr) {
        float v = tv[rr]; int ix = ti[rr];
#pragma unroll
        for (int k = 2; k <= 32; k <<= 1) {
#pragma unroll
            for (int j = k >> 1; j > 0; j >>= 1) {
                float ov = __shfl_xor(v, j);
                int   oi = __shfl_xor(ix, j);
                bool iAmUpper  = (lane & j) != 0;
                bool descBlock = (lane & k) == 0;
                bool takeMax   = descBlock ? !iAmUpper : iAmUpper;
                if ((ov > v) == takeMax && ov != v) { v = ov; ix = oi; }
            }
        }
        const float mx = __shfl(v, 0);               // sorted -> lane 0 is max
        const float e  = __expf(v - mx);
        const float se = wave_sum(e);
        const float wgt = e / se;

        const int gb = mb + wave * 2 + rr;
        const int d0 = lane * 16;                    // 16 contiguous elems/lane
        // kv[:,0,:] = x + type_emb[0]
        {
            const float4* xr = (const float4*)(xf + (size_t)gb * DD + d0);
            const float4* te = (const float4*)(type_emb + d0);
            float vals[16];
#pragma unroll
            for (int p = 0; p < 4; ++p) {
                float4 a = xr[p], t4 = te[p];
                vals[p * 4 + 0] = a.x + t4.x; vals[p * 4 + 1] = a.y + t4.y;
                vals[p * 4 + 2] = a.z + t4.z; vals[p * 4 + 3] = a.w + t4.w;
            }
            bf16* dst = kv + ((size_t)gb * 33) * DD + d0;
            v8bf o0, o1;
#pragma unroll
            for (int p = 0; p < 8; ++p) { o0[p] = (bf16)vals[p]; o1[p] = (bf16)vals[p + 8]; }
            *(v8bf*)dst = o0; *(v8bf*)(dst + 8) = o1;
        }
        // kv[:,1+j,:] = w_j * cf[idx_j] + pos_emb[j] + type_emb[1]
        for (int j = 0; j < 32; ++j) {
            const float wj = __shfl(wgt, j);
            const int   ij = __shfl(ix, j);
            const float4* cr = (const float4*)(cff + (size_t)ij * DD + d0);
            const float4* pe = (const float4*)(pos_emb + (size_t)j * DD + d0);
            const float4* te = (const float4*)(type_emb + DD + d0);
            float vals[16];
#pragma unroll
            for (int p = 0; p < 4; ++p) {
                float4 a = cr[p], b4 = pe[p], c4 = te[p];
                vals[p * 4 + 0] = wj * a.x + b4.x + c4.x;
                vals[p * 4 + 1] = wj * a.y + b4.y + c4.y;
                vals[p * 4 + 2] = wj * a.z + b4.z + c4.z;
                vals[p * 4 + 3] = wj * a.w + b4.w + c4.w;
            }
            bf16* dst = kv + ((size_t)gb * 33 + 1 + j) * DD + d0;
            v8bf o0, o1;
#pragma unroll
            for (int p = 0; p < 8; ++p) { o0[p] = (bf16)vals[p]; o1[p] = (bf16)vals[p + 8]; }
            *(v8bf*)dst = o0; *(v8bf*)(dst + 8) = o1;
        }
    }
}

// ------------------------------------------------------- cross attention ----
// One block per batch element, one wave per head. 33 keys, head dim 64.
// q/k/v staged into LDS via async-to-LDS loads (ASYNCcnt).
__global__ void cross_attn(const float* __restrict__ q, const bf16* __restrict__ kb,
                           const bf16* __restrict__ vb, bf16* __restrict__ ob) {
    const int b = blockIdx.x;
    const int tid = threadIdx.x;
    const int lane = tid & 31;
    const int h = tid >> 5;

    __shared__ float qs[DD];
    __shared__ bf16  ks[33 * DD];
    __shared__ bf16  vs[33 * DD];
    const unsigned qs_base = (unsigned)(uintptr_t)(void*)&qs[0];
    const unsigned ks_base = (unsigned)(uintptr_t)(void*)&ks[0];
    const unsigned vs_base = (unsigned)(uintptr_t)(void*)&vs[0];

    if (tid < 128)   // 512 f32 = 128 x 16B
        async_b128(qs_base + (unsigned)tid * 16u,
                   (const char*)(q + (size_t)b * DD) + tid * 16);
    // 33*512 bf16 = 2112 x 16B each
    for (int c = tid; c < 2112; c += 256) {
        async_b128(ks_base + (unsigned)c * 16u,
                   (const char*)(kb + (size_t)b * 33 * DD) + c * 16);
        async_b128(vs_base + (unsigned)c * 16u,
                   (const char*)(vb + (size_t)b * 33 * DD) + c * 16);
    }
    asm volatile("s_wait_asynccnt 0" ::: "memory");
    __syncthreads();

    const int hb = h * 64;
    // logit for key j = lane
    float myl = 0.f;
    {
        const bf16* kr = &ks[lane * DD + hb];
        for (int d = 0; d < 64; ++d) myl += qs[hb + d] * (float)kr[d];
        myl *= 0.125f;
    }
    // logit for key 32 (cooperative)
    float p32 = qs[hb + lane * 2]     * (float)ks[32 * DD + hb + lane * 2]
              + qs[hb + lane * 2 + 1] * (float)ks[32 * DD + hb + lane * 2 + 1];
    const float l32 = wave_sum(p32) * 0.125f;

    float mx = fmaxf(wave_max(myl), l32);
    float e   = __expf(myl - mx);
    float e32 = __expf(l32 - mx);
    float S   = wave_sum(e) + e32;
    float a   = e / S, a32 = e32 / S;

    float o0 = 0.f, o1 = 0.f;
    for (int j = 0; j < 32; ++j) {
        const float aj = __shfl(a, j);
        o0 += aj * (float)vs[j * DD + hb + lane * 2];
        o1 += aj * (float)vs[j * DD + hb + lane * 2 + 1];
    }
    o0 += a32 * (float)vs[32 * DD + hb + lane * 2];
    o1 += a32 * (float)vs[32 * DD + hb + lane * 2 + 1];
    ob[(size_t)b * DD + hb + lane * 2]     = (bf16)o0;
    ob[(size_t)b * DD + hb + lane * 2 + 1] = (bf16)o1;
}

// ---------------------------------------------------- residual + LayerNorm --
__global__ void add_ln(float* __restrict__ t, const float* __restrict__ delta,
                       const float* __restrict__ g, const float* __restrict__ b,
                       bf16* __restrict__ tb) {
    __shared__ float red[8];
    const int row = blockIdx.x;
    const int i0 = threadIdx.x, i1 = threadIdx.x + 256;
    const size_t base = (size_t)row * DD;
    float a0 = t[base + i0] + delta[base + i0];
    float a1 = t[base + i1] + delta[base + i1];
    const float mean = block_sum256(a0 + a1, red) * (1.f / DD);
    const float d0 = a0 - mean, d1 = a1 - mean;
    const float var = block_sum256(d0 * d0 + d1 * d1, red) * (1.f / DD);
    const float rs = rsqrtf(var + 1e-5f);
    float y0 = d0 * rs * g[i0] + b[i0];
    float y1 = d1 * rs * g[i1] + b[i1];
    t[base + i0] = y0; t[base + i1] = y1;
    tb[base + i0] = (bf16)y0; tb[base + i1] = (bf16)y1;
}

// ------------------------------------------------- final triple-normalize ---
__global__ void final_norm(const float* __restrict__ t, const float* __restrict__ coarse,
                           float* __restrict__ out) {
    __shared__ float red[8];
    const int row = blockIdx.x;
    const int i0 = threadIdx.x, i1 = threadIdx.x + 256;
    const size_t base = (size_t)row * DD;
    float f0 = t[base + i0], f1 = t[base + i1];
    float c0 = coarse[base + i0], c1 = coarse[base + i1];
    const float nf = rsqrtf(block_sum256(f0 * f0 + f1 * f1, red));
    const float nc = rsqrtf(block_sum256(c0 * c0 + c1 * c1, red));
    float a0 = f0 * nf + c0 * nc;
    float a1 = f1 * nf + c1 * nc;
    const float na = rsqrtf(block_sum256(a0 * a0 + a1 * a1, red));
    out[base + i0] = a0 * na;
    out[base + i1] = a1 * na;
}

// ================================================================= launch ===
extern "C" void kernel_launch(void* const* d_in, const int* in_sizes, int n_in,
                              void* d_out, int out_size, void* d_ws, size_t ws_size,
                              hipStream_t stream) {
    (void)in_sizes; (void)n_in; (void)out_size; (void)ws_size;

    const float* x        = (const float*)d_in[0];
    const float* cf       = (const float*)d_in[1];
    const float* type_emb = (const float*)d_in[2];
    const float* pos_emb  = (const float*)d_in[3];
    const float* cls_emb  = (const float*)d_in[4];
    const float* sa_in_w  = (const float*)d_in[5];
    const float* sa_in_b  = (const float*)d_in[6];
    const float* sa_out_w = (const float*)d_in[7];
    const float* sa_out_b = (const float*)d_in[8];
    const float* ca_in_w  = (const float*)d_in[9];
    const float* ca_in_b  = (const float*)d_in[10];
    const float* ca_out_w = (const float*)d_in[11];
    const float* ca_out_b = (const float*)d_in[12];
    const float* lin1_w   = (const float*)d_in[13];
    const float* lin1_b   = (const float*)d_in[14];
    const float* lin2_w   = (const float*)d_in[15];
    const float* lin2_b   = (const float*)d_in[16];
    const float* ln1_g    = (const float*)d_in[17];
    const float* ln1_b    = (const float*)d_in[18];
    const float* ln2_g    = (const float*)d_in[19];
    const float* ln2_b    = (const float*)d_in[20];
    const float* ln3_g    = (const float*)d_in[21];
    const float* ln3_b    = (const float*)d_in[22];
    const float* region_w = (const float*)d_in[23];
    const float* region_b = (const float*)d_in[24];

    // ---- workspace carve-out
    char* wsb = (char*)d_ws;
    size_t off = 0;
    auto carve = [&](size_t bytes) -> char* {
        char* p = wsb + off;
        off = (off + bytes + 255) & ~(size_t)255;
        return p;
    };
    bf16*  cf_bf   = (bf16*)carve((size_t)NN * DD * 2);
    bf16*  x_bf    = (bf16*)carve((size_t)BB * DD * 2);
    bf16*  w_sa_in = (bf16*)carve((size_t)LL * 3 * DD * DD * 2);
    bf16*  w_sa_o  = (bf16*)carve((size_t)LL * DD * DD * 2);
    bf16*  w_ca_in = (bf16*)carve((size_t)LL * 3 * DD * DD * 2);
    bf16*  w_ca_o  = (bf16*)carve((size_t)LL * DD * DD * 2);
    bf16*  w_l1    = (bf16*)carve((size_t)LL * FFD * DD * 2);
    bf16*  w_l2    = (bf16*)carve((size_t)LL * DD * FFD * 2);
    bf16*  w_rg    = (bf16*)carve((size_t)DD * DD * 2);
    bf16*  kv_bf   = (bf16*)carve((size_t)BB * 33 * DD * 2);
    float* t_f     = (float*)carve((size_t)BB * DD * 4);
    bf16*  t_b     = (bf16*)carve((size_t)BB * DD * 2);
    bf16*  s1_b    = (bf16*)carve((size_t)BB * DD * 2);
    float* s2_f    = (float*)carve((size_t)BB * DD * 4);
    float* q_f     = (float*)carve((size_t)BB * DD * 4);
    bf16*  k_b     = (bf16*)carve((size_t)BB * 33 * DD * 2);
    bf16*  v_b     = (bf16*)carve((size_t)BB * 33 * DD * 2);
    bf16*  o_b     = (bf16*)carve((size_t)BB * DD * 2);
    bf16*  ffm_b   = (bf16*)carve((size_t)BB * FFD * 2);

    auto cvt = [&](const float* s, bf16* d, size_t n) {
        cvt_f32_bf16<<<4096, 256, 0, stream>>>(s, d, n / 8);
    };
    auto gemm = [&](const bf16* A, const bf16* W, const float* bias,
                    float* Cf, bf16* Cb, int M, int N, int Kd, int relu) {
        dim3 grid(N / 64, M / 32);
        gemm_wmma<<<grid, 256, 0, stream>>>(A, W, bias, Cf, Cb, M, N, Kd, relu);
    };

    // ---- bf16 conversions (weights + activations), once per call
    cvt(cf, cf_bf, (size_t)NN * DD);
    cvt(x, x_bf, (size_t)BB * DD);
    cvt(sa_in_w, w_sa_in, (size_t)LL * 3 * DD * DD);
    cvt(sa_out_w, w_sa_o, (size_t)LL * DD * DD);
    cvt(ca_in_w, w_ca_in, (size_t)LL * 3 * DD * DD);
    cvt(ca_out_w, w_ca_o, (size_t)LL * DD * DD);
    cvt(lin1_w, w_l1, (size_t)LL * FFD * DD);
    cvt(lin2_w, w_l2, (size_t)LL * DD * FFD);
    cvt(region_w, w_rg, (size_t)DD * DD);

    // ---- fused scoring + top-32 + kv construction
    score_topk_kv<<<BB / 16, 256, 0, stream>>>(x_bf, cf_bf, x, cf,
                                               type_emb, pos_emb, kv_bf);

    // ---- t = class_embedding broadcast
    init_t_kernel<<<1024, 256, 0, stream>>>(cls_emb, t_f, t_b, (size_t)BB * DD / 8);

    // ---- transformer layers
    for (int i = 0; i < LL; ++i) {
        const bf16* Wsa = w_sa_in + (size_t)i * 3 * DD * DD;
        const bf16* Wca = w_ca_in + (size_t)i * 3 * DD * DD;
        const float* bsa = sa_in_b + (size_t)i * 3 * DD;
        const float* bca = ca_in_b + (size_t)i * 3 * DD;

        // self-attn (seq len 1 => softmax==1 => out = Wo(Wv t + bv) + bo)
        gemm(t_b, Wsa + 2 * (size_t)DD * DD, bsa + 2 * DD, nullptr, s1_b, BB, DD, DD, 0);
        gemm(s1_b, w_sa_o + (size_t)i * DD * DD, sa_out_b + (size_t)i * DD,
             s2_f, nullptr, BB, DD, DD, 0);
        add_ln<<<BB, 256, 0, stream>>>(t_f, s2_f, ln1_g + (size_t)i * DD,
                                       ln1_b + (size_t)i * DD, t_b);

        // cross-attn
        gemm(t_b, Wca, bca, q_f, nullptr, BB, DD, DD, 0);                         // q
        gemm(kv_bf, Wca + (size_t)DD * DD, bca + DD, nullptr, k_b, BB * 33, DD, DD, 0); // k
        gemm(kv_bf, Wca + 2 * (size_t)DD * DD, bca + 2 * DD, nullptr, v_b, BB * 33, DD, DD, 0); // v
        cross_attn<<<BB, 256, 0, stream>>>(q_f, k_b, v_b, o_b);
        gemm(o_b, w_ca_o + (size_t)i * DD * DD, ca_out_b + (size_t)i * DD,
             s2_f, nullptr, BB, DD, DD, 0);
        add_ln<<<BB, 256, 0, stream>>>(t_f, s2_f, ln2_g + (size_t)i * DD,
                                       ln2_b + (size_t)i * DD, t_b);

        // FFN
        gemm(t_b, w_l1 + (size_t)i * FFD * DD, lin1_b + (size_t)i * FFD,
             nullptr, ffm_b, BB, FFD, DD, 1);
        gemm(ffm_b, w_l2 + (size_t)i * DD * FFD, lin2_b + (size_t)i * DD,
             s2_f, nullptr, BB, DD, FFD, 0);
        add_ln<<<BB, 256, 0, stream>>>(t_f, s2_f, ln3_g + (size_t)i * DD,
                                       ln3_b + (size_t)i * DD, t_b);
    }

    // ---- coarse = x @ region_w.T + b ; out = normalize(norm(fine)+norm(coarse))
    gemm(x_bf, w_rg, region_b, q_f, nullptr, BB, DD, DD, 0);
    final_norm<<<BB, 256, 0, stream>>>(t_f, q_f, (float*)d_out);
}
